// SparseConvNeXtBlock3d_858993459501
// MI455X (gfx1250) — compile-verified, matrix operand order checked by reading, not run on hardware
//
#include <hip/hip_runtime.h>
#include <hip/hip_bf16.h>

#define GDIM 66
#define GVOX (GDIM * GDIM * GDIM)
#define NPTS 65536
#define C_ 512
#define H_ 2048
#define EPS_ 1e-6f

typedef __attribute__((ext_vector_type(16))) __bf16       v16bf;
typedef __attribute__((ext_vector_type(8)))  float        v8f;
typedef __attribute__((ext_vector_type(4)))  unsigned int uiv4;

union BFrag { v16bf v; uiv4 q[2]; };

__device__ __forceinline__ unsigned short f2bf(float f) {
  union { float f; unsigned u; } v; v.f = f;
  unsigned r = v.u + 0x7FFFu + ((v.u >> 16) & 1u);   // round-to-nearest-even
  return (unsigned short)(r >> 16);
}

__device__ __forceinline__ v8f zero8() {
  v8f z;
#pragma unroll
  for (int i = 0; i < 8; ++i) z[i] = 0.0f;
  return z;
}

__device__ __forceinline__ v8f wmma_bf16(const BFrag& a, const BFrag& b, v8f c) {
  return __builtin_amdgcn_wmma_f32_16x16x32_bf16(false, a.v, false, b.v,
                                                 (short)0, c, false, false);
}

// ---- CDNA5 async global->LDS copies (ASYNCcnt-tracked) --------------------
__device__ __forceinline__ void cp_async_b128(void* lds, const void* gptr) {
  unsigned l = (unsigned)(unsigned long long)lds;
  asm volatile("global_load_async_to_lds_b128 %0, %1, off"
               :: "v"(l), "v"(gptr) : "memory");
}
__device__ __forceinline__ void cp_async_b64(void* lds, const void* gptr) {
  unsigned l = (unsigned)(unsigned long long)lds;
  asm volatile("global_load_async_to_lds_b64 %0, %1, off"
               :: "v"(l), "v"(gptr) : "memory");
}
#define WAIT_ASYNC(n) asm volatile("s_wait_asynccnt " #n ::: "memory")

// ---------------------------------------------------------------- grid build
__global__ void k_grid_init(int* __restrict__ grid) {
  int i = blockIdx.x * 256 + threadIdx.x;
  if (i < GVOX) grid[i] = -1;
}

__global__ void k_grid_scatter(const int* __restrict__ coords, int* __restrict__ grid) {
  int i = blockIdx.x * 256 + threadIdx.x;
  if (i < NPTS) {
    int x = coords[i * 3 + 0] + 1;
    int y = coords[i * 3 + 1] + 1;
    int z = coords[i * 3 + 2] + 1;
    grid[(x * GDIM + y) * GDIM + z] = i;
  }
}

// ------------------------------------------- one-time convert / swizzle pass
// feats f32 [N][512] -> bf16, plus one zero row at index NPTS
__global__ void k_cvt_feats(const float* __restrict__ f, unsigned short* __restrict__ o) {
  if (blockIdx.x == (NPTS * C_) / 1024) {           // zero row
    if (threadIdx.x < 128) {
      size_t b = (size_t)NPTS * C_ + threadIdx.x * 4;
      o[b] = 0; o[b + 1] = 0; o[b + 2] = 0; o[b + 3] = 0;
    }
    return;
  }
  size_t b = ((size_t)blockIdx.x * 256 + threadIdx.x) * 4;
  float4 v = *(const float4*)(f + b);
  o[b + 0] = f2bf(v.x); o[b + 1] = f2bf(v.y);
  o[b + 2] = f2bf(v.z); o[b + 3] = f2bf(v.w);
}

// conv_w f32 [27][512][512] -> bf16 tiles [27*16][512 col][32 k]
__global__ void k_swz_convw(const float* __restrict__ w, unsigned short* __restrict__ o) {
  size_t idx = (size_t)blockIdx.x * 256 + threadIdx.x;   // 27*512*512
  int col = (int)(idx & 511);
  size_t t  = idx >> 9;                                  // tap*512 + k
  int k = (int)(t & 511), tap = (int)(t >> 9);
  int kc = k >> 5, kk = k & 31;
  o[(((size_t)(tap * 16 + kc) * C_) + col) * 32 + kk] = f2bf(w[idx]);
}

// w1 f32 [512][2048] -> bf16 tiles [16 hb][16 kc][128 col][32 k]
__global__ void k_swz_w1(const float* __restrict__ w, unsigned short* __restrict__ o) {
  size_t idx = (size_t)blockIdx.x * 256 + threadIdx.x;   // 512*2048
  int h = (int)(idx & 2047);
  int k = (int)(idx >> 11);
  int hb = h >> 7, col = h & 127, kc = k >> 5, kk = k & 31;
  o[(((size_t)(hb * 16 + kc) * 128) + col) * 32 + kk] = f2bf(w[idx]);
}

// w2 f32 [2048][512] -> bf16 tiles [64 kc][512 col][32 k]
__global__ void k_swz_w2(const float* __restrict__ w, unsigned short* __restrict__ o) {
  size_t idx = (size_t)blockIdx.x * 256 + threadIdx.x;   // 2048*512
  int col = (int)(idx & 511);
  int k   = (int)(idx >> 9);
  int kc = k >> 5, kk = k & 31;
  o[(((size_t)kc * C_) + col) * 32 + kk] = f2bf(w[idx]);
}

// ------------------------------------------------- sparse conv + bias + LN
// block: 32 points x 512 out channels, 8 waves; wave = 16 rows x 128 cols
// 432 K-chunks (27 taps x 16), double-buffered async pipeline
__global__ __launch_bounds__(256) void k_conv_ln(
    const unsigned short* __restrict__ feats_bf, const int* __restrict__ coords,
    const unsigned short* __restrict__ cw_bf, const float* __restrict__ conv_b,
    const float* __restrict__ ln_g, const float* __restrict__ ln_b,
    const int* __restrict__ grid, unsigned short* __restrict__ hn) {
  __shared__ int            s_nbr[27 * 32];
  __shared__ unsigned short s_a[2][32 * 32];
  __shared__ unsigned short s_w[2][C_ * 32];
  __shared__ float          s_h[32 * C_];
  __shared__ float          s_red[256];
  __shared__ float          s_mu[32], s_rs[32];

  const int tid   = threadIdx.x;
  const int lane  = tid & 31;
  const int wv    = tid >> 5;
  const int wm    = wv & 1;
  const int wn    = wv >> 1;
  const int khalf = lane >> 4;
  const int l16   = lane & 15;
  const int rowbase = blockIdx.x * 32;
  const int r8 = tid >> 3, p8 = tid & 7;

  // ---- neighbor tables for all 27 taps (invalid -> zero row NPTS)
  for (int idx = tid; idx < 27 * 32; idx += 256) {
    int tap = idx >> 5, r = idx & 31;
    int g  = rowbase + r;
    int dx = tap / 9, dy = (tap / 3) % 3, dz = tap % 3;
    int x = coords[g * 3 + 0] + dx;
    int y = coords[g * 3 + 1] + dy;
    int z = coords[g * 3 + 2] + dz;
    int nb = grid[(x * GDIM + y) * GDIM + z];
    s_nbr[idx] = (nb < 0) ? NPTS : nb;
  }
  __syncthreads();

  // per-chunk async stage: 1 b64 (gathered A) + 8 b128 (W tile) per thread
  auto issue = [&](int chunk, int buf) {
    int tap = chunk >> 4, kc = chunk & 15;
    int nb  = s_nbr[tap * 32 + r8];
    cp_async_b64(&s_a[buf][r8 * 32 + p8 * 4],
                 feats_bf + (size_t)nb * C_ + kc * 32 + p8 * 4);
    const unsigned short* wsrc = cw_bf + (size_t)chunk * (C_ * 32);
#pragma unroll
    for (int i = 0; i < 8; ++i)
      cp_async_b128(&s_w[buf][(tid + i * 256) * 8], wsrc + (tid + i * 256) * 8);
  };

  v8f acc[8];
#pragma unroll
  for (int i = 0; i < 8; ++i) acc[i] = zero8();

  issue(0, 0);
  for (int c = 0; c < 432; ++c) {
    if (c + 1 < 432) { issue(c + 1, (c + 1) & 1); WAIT_ASYNC(9); }
    else             { WAIT_ASYNC(0); }
    __syncthreads();

    const unsigned short* sa = s_a[c & 1];
    const unsigned short* sw = s_w[c & 1];
    BFrag a;
    const unsigned short* ap = sa + (wm * 16 + l16) * 32 + khalf * 8;
    a.q[0] = *(const uiv4*)(ap);
    a.q[1] = *(const uiv4*)(ap + 16);
#pragma unroll
    for (int nt = 0; nt < 8; ++nt) {
      BFrag b;
      const unsigned short* bp = sw + (wn * 128 + nt * 16 + l16) * 32 + khalf * 8;
      b.q[0] = *(const uiv4*)(bp);
      b.q[1] = *(const uiv4*)(bp + 16);
      acc[nt] = wmma_bf16(a, b, acc[nt]);
    }
    __syncthreads();
  }

  // ---- dump accumulators + bias (C layout: M = r + 8*khalf)
#pragma unroll
  for (int nt = 0; nt < 8; ++nt) {
    int col    = wn * 128 + nt * 16 + l16;
    float bias = conv_b[col];
#pragma unroll
    for (int r = 0; r < 8; ++r) {
      int m = wm * 16 + khalf * 8 + r;
      s_h[m * C_ + col] = acc[nt][r] + bias;
    }
  }
  __syncthreads();

  // ---- LayerNorm over C=512 per row
  {
    int row = tid >> 3, j = tid & 7;
    float p = 0.f;
    for (int c = j; c < C_; c += 8) p += s_h[row * C_ + c];
    s_red[tid] = p;
    __syncthreads();
    if (tid < 32) {
      float m = 0.f;
#pragma unroll
      for (int i = 0; i < 8; ++i) m += s_red[tid * 8 + i];
      s_mu[tid] = m * (1.0f / C_);
    }
    __syncthreads();
    float mu = s_mu[row];
    p = 0.f;
    for (int c = j; c < C_; c += 8) {
      float d = s_h[row * C_ + c] - mu;
      p += d * d;
    }
    s_red[tid] = p;
    __syncthreads();
    if (tid < 32) {
      float vs = 0.f;
#pragma unroll
      for (int i = 0; i < 8; ++i) vs += s_red[tid * 8 + i];
      s_rs[tid] = rsqrtf(vs * (1.0f / C_) + EPS_);
    }
    __syncthreads();
    for (int idx = tid; idx < 32 * C_; idx += 256) {
      int row2 = idx >> 9, col = idx & (C_ - 1);
      float v = (s_h[idx] - s_mu[row2]) * s_rs[row2] * ln_g[col] + ln_b[col];
      hn[(size_t)rowbase * C_ + idx] = f2bf(v);
    }
  }
}

// --------------------------------------------- MLP: silu(hn@w1+b1)@w2+b2+feats
// block: 16 points; activations kept in LDS; async double-buffered weights
__global__ __launch_bounds__(256) void k_mlp(
    const unsigned short* __restrict__ hn, const float* __restrict__ feats,
    const unsigned short* __restrict__ w1_bf, const float* __restrict__ b1,
    const unsigned short* __restrict__ w2_bf, const float* __restrict__ b2,
    float* __restrict__ out) {
  __shared__ unsigned short s_hn[16 * C_];       // 16 KB
  __shared__ unsigned short s_a1[16 * H_];       // 64 KB
  __shared__ unsigned short s_wt[2][C_ * 32];    // 2 x 32 KB

  const int tid   = threadIdx.x;
  const int lane  = tid & 31;
  const int wv    = tid >> 5;
  const int khalf = lane >> 4;
  const int l16   = lane & 15;
  const int rowbase = blockIdx.x * 16;

  // ---- stage the 16 hidden rows once (async, contiguous bf16)
  {
    const unsigned short* src = hn + (size_t)rowbase * C_;
#pragma unroll
    for (int i = 0; i < 4; ++i)
      cp_async_b128(&s_hn[(tid + i * 256) * 8], src + (tid + i * 256) * 8);
  }

  // ---- GEMM1: flat 256 tiles of (16 x 16cols x 32k); 1 tile per wave
  auto issue1 = [&](int t, int buf) {
    const unsigned short* src = w1_bf + (size_t)t * (128 * 32);
#pragma unroll
    for (int i = 0; i < 2; ++i)
      cp_async_b128(&s_wt[buf][(tid + i * 256) * 8], src + (tid + i * 256) * 8);
  };

  v8f a1 = zero8();
  issue1(0, 0);
  WAIT_ASYNC(0);           // hn + first tile resident
  __syncthreads();
  for (int t = 0; t < 256; ++t) {
    if (t + 1 < 256) { issue1(t + 1, (t + 1) & 1); WAIT_ASYNC(2); }
    else             { WAIT_ASYNC(0); }
    __syncthreads();

    const int k0 = (t & 15) * 32;
    BFrag a;
    const unsigned short* ap = s_hn + l16 * C_ + k0 + khalf * 8;
    a.q[0] = *(const uiv4*)(ap);
    a.q[1] = *(const uiv4*)(ap + 16);
    BFrag b;
    const unsigned short* bp = s_wt[t & 1] + (wv * 16 + l16) * 32 + khalf * 8;
    b.q[0] = *(const uiv4*)(bp);
    b.q[1] = *(const uiv4*)(bp + 16);
    a1 = wmma_bf16(a, b, a1);

    if ((t & 15) == 15) {                 // end of K for this H chunk: SiLU
      int   hbase = (t >> 4) * 128;
      int   colg  = hbase + wv * 16 + l16;
      float bb    = b1[colg];
#pragma unroll
      for (int r = 0; r < 8; ++r) {
        int   m  = khalf * 8 + r;
        float x  = a1[r] + bb;
        float sx = x / (1.0f + __expf(-x));
        s_a1[m * H_ + colg] = f2bf(sx);
      }
      a1 = zero8();
    }
    __syncthreads();
  }

  // ---- GEMM2: 16 x 512, K = 2048; wave = 64 cols (4 tiles), 64 K-chunks
  auto issue2 = [&](int t, int buf) {
    const unsigned short* src = w2_bf + (size_t)t * (C_ * 32);
#pragma unroll
    for (int i = 0; i < 8; ++i)
      cp_async_b128(&s_wt[buf][(tid + i * 256) * 8], src + (tid + i * 256) * 8);
  };

  v8f acc2[4];
#pragma unroll
  for (int i = 0; i < 4; ++i) acc2[i] = zero8();

  issue2(0, 0);
  for (int t = 0; t < 64; ++t) {
    if (t + 1 < 64) { issue2(t + 1, (t + 1) & 1); WAIT_ASYNC(8); }
    else            { WAIT_ASYNC(0); }
    __syncthreads();

    BFrag a;
    const unsigned short* ap = s_a1 + l16 * H_ + t * 32 + khalf * 8;
    a.q[0] = *(const uiv4*)(ap);
    a.q[1] = *(const uiv4*)(ap + 16);
#pragma unroll
    for (int nt = 0; nt < 4; ++nt) {
      BFrag b;
      const unsigned short* bp = s_wt[t & 1] + (wv * 64 + nt * 16 + l16) * 32 + khalf * 8;
      b.q[0] = *(const uiv4*)(bp);
      b.q[1] = *(const uiv4*)(bp + 16);
      acc2[nt] = wmma_bf16(a, b, acc2[nt]);
    }
    __syncthreads();
  }

  // ---- epilogue: + b2 + residual
#pragma unroll
  for (int nt = 0; nt < 4; ++nt) {
    int   col = wv * 64 + nt * 16 + l16;
    float bb  = b2[col];
#pragma unroll
    for (int r = 0; r < 8; ++r) {
      int    m = khalf * 8 + r;
      size_t o = (size_t)(rowbase + m) * C_ + col;
      out[o] = acc2[nt][r] + bb + feats[o];
    }
  }
}

// ---------------------------------------------------------------- launcher
extern "C" void kernel_launch(void* const* d_in, const int* in_sizes, int n_in,
                              void* d_out, int out_size, void* d_ws, size_t ws_size,
                              hipStream_t stream) {
  const float* feats  = (const float*)d_in[0];
  const int*   coords = (const int*)d_in[1];
  const float* conv_w = (const float*)d_in[2];
  const float* conv_b = (const float*)d_in[3];
  const float* ln_g   = (const float*)d_in[4];
  const float* ln_b   = (const float*)d_in[5];
  const float* w1     = (const float*)d_in[6];
  const float* b1     = (const float*)d_in[7];
  const float* w2     = (const float*)d_in[8];
  const float* b2     = (const float*)d_in[9];
  float*       out    = (float*)d_out;

  char* ws = (char*)d_ws;
  const size_t MB = 1u << 20;
  int*            grid     = (int*)(ws);                    //   1.15 MB
  unsigned short* feats_bf = (unsigned short*)(ws + 2 * MB);   //  64 MB (+zero row)
  unsigned short* hn       = (unsigned short*)(ws + 70 * MB);  //  64 MB
  unsigned short* cw_bf    = (unsigned short*)(ws + 134 * MB); // 13.5 MB
  unsigned short* w1_bf    = (unsigned short*)(ws + 148 * MB); //   2 MB
  unsigned short* w2_bf    = (unsigned short*)(ws + 150 * MB); //   2 MB

  k_grid_init<<<(GVOX + 255) / 256, 256, 0, stream>>>(grid);
  k_grid_scatter<<<NPTS / 256, 256, 0, stream>>>(coords, grid);
  k_cvt_feats<<<(NPTS * C_) / 1024 + 1, 256, 0, stream>>>(feats, feats_bf);
  k_swz_convw<<<(27 * C_ * C_) / 256, 256, 0, stream>>>(conv_w, cw_bf);
  k_swz_w1<<<(C_ * H_) / 256, 256, 0, stream>>>(w1, w1_bf);
  k_swz_w2<<<(H_ * C_) / 256, 256, 0, stream>>>(w2, w2_bf);
  k_conv_ln<<<NPTS / 32, 256, 0, stream>>>(feats_bf, coords, cw_bf, conv_b,
                                           ln_g, ln_b, grid, hn);
  k_mlp<<<NPTS / 16, 256, 0, stream>>>(hn, feats, w1_bf, b1, w2_bf, b2, out);
}
